// DynamicSparsityRouter_89945205113688
// MI455X (gfx1250) — matrix operand order repaired and checked by reading, not compiled, not run on hardware
//
#include <hip/hip_runtime.h>
#include <hip/hip_bf16.h>

#define HDIM 2048
#define NEXP 64
#define D1   512
#define NTOK 16384
#define KMIN 4.0f
#define KMAX 12.0f
#define BASEK 8.0f
#define AUXW 0.01f
#define ENTW 0.001f
#define KC   128                 // K-chunk staged to LDS by TDM

typedef __attribute__((ext_vector_type(16))) __bf16 v16bf;
typedef __attribute__((ext_vector_type(8)))  __bf16 v8bf;
typedef __attribute__((ext_vector_type(8)))  float  v8f;
typedef __attribute__((ext_vector_type(4)))  unsigned int u32x4;
typedef __attribute__((ext_vector_type(8)))  int i32x8;
typedef __attribute__((ext_vector_type(4)))  int i32x4;

#if __has_builtin(__builtin_amdgcn_tensor_load_to_lds) && __has_builtin(__builtin_amdgcn_s_wait_tensorcnt)
#define HAVE_TDM 1
#else
#define HAVE_TDM 0
#endif

#define WMMA_BF16(a, b, c) \
  __builtin_amdgcn_wmma_f32_16x16x32_bf16(false, (a), false, (b), (short)0, (c), false, false)

// ---------------------------------------------------------------- helpers --
static __device__ __forceinline__ v16bf cvt16(float4 a0, float4 a1, float4 a2, float4 a3) {
  v16bf f;
  f[0]=(__bf16)a0.x; f[1]=(__bf16)a0.y; f[2]=(__bf16)a0.z; f[3]=(__bf16)a0.w;
  f[4]=(__bf16)a1.x; f[5]=(__bf16)a1.y; f[6]=(__bf16)a1.z; f[7]=(__bf16)a1.w;
  f[8]=(__bf16)a2.x; f[9]=(__bf16)a2.y; f[10]=(__bf16)a2.z; f[11]=(__bf16)a2.w;
  f[12]=(__bf16)a3.x; f[13]=(__bf16)a3.y; f[14]=(__bf16)a3.z; f[15]=(__bf16)a3.w;
  return f;
}

// A fragment (16x32 bf16 MxK) from an f32 row (fallback path).
static __device__ __forceinline__ v16bf load_a_frag_f32(const float* __restrict__ rowp, int lane, int k0) {
  const float* hp = rowp + k0 + (((lane >> 4) & 1) << 4);
  return cvt16(*reinterpret_cast<const float4*>(hp),
               *reinterpret_cast<const float4*>(hp + 4),
               *reinterpret_cast<const float4*>(hp + 8),
               *reinterpret_cast<const float4*>(hp + 12));
}
static __device__ __forceinline__ v16bf load_b_frag_f32(const float* __restrict__ wp) {
  return cvt16(*reinterpret_cast<const float4*>(wp),
               *reinterpret_cast<const float4*>(wp + 4),
               *reinterpret_cast<const float4*>(wp + 8),
               *reinterpret_cast<const float4*>(wp + 12));
}

// A fragment from a packed-bf16 row: two contiguous 16B halves.
static __device__ __forceinline__ v16bf load_a_frag_bf(const __bf16* rowp, int g8, int k0) {
  v8bf lo = *reinterpret_cast<const v8bf*>(rowp + k0 + 2 * g8);
  v8bf hi = *reinterpret_cast<const v8bf*>(rowp + k0 + 2 * g8 + 8);
  return __builtin_shufflevector(lo, hi, 0,1,2,3,4,5,6,7,8,9,10,11,12,13,14,15);
}
// B fragment: 16 contiguous bf16 (32B, 32B-aligned).
static __device__ __forceinline__ v16bf load_b_frag_bf(const __bf16* wp) {
  return *reinterpret_cast<const v16bf*>(wp);
}

#if HAVE_TDM
// Issue a 2D TDM load: tile_rows x tile_k bf16 elements, row stride = stride
// elements, from global (packed bf16) into LDS byte offset lds_off.
// D# bitfields per CDNA5 ISA.
static __device__ __forceinline__ void tdm_load_2d(unsigned lds_off, const __bf16* gaddr,
                                                   unsigned tile_k, unsigned tile_rows,
                                                   unsigned stride) {
  unsigned long long ga = (unsigned long long)(uintptr_t)gaddr;
  u32x4 g0;
  g0[0] = 1u;                                               // count=1 (valid user D#)
  g0[1] = lds_off;                                          // lds_addr (bytes)
  g0[2] = (unsigned)(ga & 0xffffffffu);                     // global_addr[31:0]
  g0[3] = (unsigned)((ga >> 32) & 0x01ffffffu) | 0x80000000u; // addr[56:32] | type=2
  i32x8 g1;
  g1[0] = (int)(1u << 16);                                  // data_size=1 -> 2 bytes
  g1[1] = (int)((tile_k & 0xffffu) << 16);                  // tensor_dim0[15:0] = tile_k
  g1[2] = (int)(((tile_k >> 16) & 0xffffu) | ((tile_rows & 0xffffu) << 16)); // td0 hi | td1 lo
  g1[3] = (int)(((tile_rows >> 16) & 0xffffu) | ((tile_k & 0xffffu) << 16)); // td1 hi | tile_dim0
  g1[4] = (int)(tile_rows & 0xffffu);                       // tile_dim1 (tile_dim2=0)
  g1[5] = (int)stride;                                      // tensor_dim0_stride[31:0]
  g1[6] = 0;
  g1[7] = 0;
  i32x4 z = {0, 0, 0, 0};
#if __clang_major__ >= 23
  i32x8 z8 = {0, 0, 0, 0, 0, 0, 0, 0};
  __builtin_amdgcn_tensor_load_to_lds(g0, g1, z, z, z8, 0);
#else
  __builtin_amdgcn_tensor_load_to_lds(g0, g1, z, z, 0);
#endif
}
#endif

// ----------------------------------------------------------- f32 -> bf16 ---
__global__ __launch_bounds__(256) void cvt_bf16_kernel(const float* __restrict__ src,
                                                       __bf16* __restrict__ dst, int n8) {
  int i = blockIdx.x * 256 + threadIdx.x;
  if (i < n8) {
    const float4* s = reinterpret_cast<const float4*>(src) + 2 * (size_t)i;
    float4 a = s[0], b = s[1];
    v8bf o;
    o[0]=(__bf16)a.x; o[1]=(__bf16)a.y; o[2]=(__bf16)a.z; o[3]=(__bf16)a.w;
    o[4]=(__bf16)b.x; o[5]=(__bf16)b.y; o[6]=(__bf16)b.z; o[7]=(__bf16)b.w;
    reinterpret_cast<v8bf*>(dst)[i] = o;
  }
}

// ------------------------- gate logits (packed-bf16 fast path) -------------
__global__ __launch_bounds__(128) void gate_logits_bf_kernel(
    const __bf16* __restrict__ hbf, const __bf16* __restrict__ gbf,
    float* __restrict__ logits) {
  int lane = threadIdx.x & 31;
  int wave = threadIdx.x >> 5;
  int row0 = (blockIdx.x * 4 + wave) * 16;
  int g8   = ((lane >> 4) & 1) << 3;
  int g16  = ((lane >> 4) & 1) << 4;
  int ncol = lane & 15;
  const __bf16* arow = hbf + (size_t)(row0 + (lane & 15)) * HDIM;
  const __bf16* b0p = gbf + (size_t)(ncol) * HDIM + g16;
  const __bf16* b1p = gbf + (size_t)(16 + ncol) * HDIM + g16;
  const __bf16* b2p = gbf + (size_t)(32 + ncol) * HDIM + g16;
  const __bf16* b3p = gbf + (size_t)(48 + ncol) * HDIM + g16;
  v8f c0 = {}, c1 = {}, c2 = {}, c3 = {};
  for (int k0 = 0; k0 < HDIM; k0 += 32) {
    __builtin_prefetch(arow + k0 + 512, 0, 1);
    v16bf a = load_a_frag_bf(arow, g8, k0);
    c0 = WMMA_BF16(a, load_b_frag_bf(b0p + k0), c0);
    c1 = WMMA_BF16(a, load_b_frag_bf(b1p + k0), c1);
    c2 = WMMA_BF16(a, load_b_frag_bf(b2p + k0), c2);
    c3 = WMMA_BF16(a, load_b_frag_bf(b3p + k0), c3);
  }
  int mhi = ((lane >> 4) & 1) << 3;
#pragma unroll
  for (int r = 0; r < 8; ++r) {
    float* orow = logits + (size_t)(row0 + mhi + r) * NEXP + ncol;
    orow[0]  = c0[r];
    orow[16] = c1[r];
    orow[32] = c2[r];
    orow[48] = c3[r];
  }
}

// ---------------- difficulty head (bf16 + TDM-staged A panel) --------------
__global__ __launch_bounds__(128) void difficulty_bf_kernel(
    const __bf16* __restrict__ hbf, const __bf16* __restrict__ wbf1,
    const float* __restrict__ b1, const float* __restrict__ w2,
    const float* __restrict__ b2, float* __restrict__ difficulty) {
#if HAVE_TDM
  __shared__ __align__(16) __bf16 lds_a[2][64 * KC];   // 2 x 16KB double buffer
#endif
  int lane = threadIdx.x & 31;
  int wave = threadIdx.x >> 5;
  int row0 = blockIdx.x * 64 + wave * 16;
  int g8   = ((lane >> 4) & 1) << 3;
  int g16  = ((lane >> 4) & 1) << 4;
  int ncol = lane & 15;
  const __bf16* arow_g = hbf + (size_t)(row0 + (lane & 15)) * HDIM;
  (void)arow_g;

  float acc[8];
#pragma unroll
  for (int r = 0; r < 8; ++r) acc[r] = 0.f;

  for (int grp = 0; grp < 4; ++grp) {        // 8 d-tiles of 16 neurons per group
    v8f c[8] = {};
    const __bf16* brow[8];
#pragma unroll
    for (int t = 0; t < 8; ++t)
      brow[t] = wbf1 + (size_t)((grp * 8 + t) * 16 + ncol) * HDIM + g16;

#if HAVE_TDM
    unsigned lds0 = (unsigned)(uintptr_t)&lds_a[0][0];
    unsigned lds1 = (unsigned)(uintptr_t)&lds_a[1][0];
    const __bf16* panel = hbf + (size_t)(blockIdx.x * 64) * HDIM;
    if (wave == 0)
      tdm_load_2d(lds0, panel, KC, 64, HDIM);
    for (int kc = 0; kc < HDIM / KC; ++kc) {
      int buf = kc & 1;
      if (wave == 0) __builtin_amdgcn_s_wait_tensorcnt(0);
      __syncthreads();
      if (wave == 0 && (kc + 1) < HDIM / KC)
        tdm_load_2d(buf ? lds0 : lds1, panel + (kc + 1) * KC, KC, 64, HDIM);
      const __bf16* lrow = &lds_a[buf][(wave * 16 + (lane & 15)) * KC];
#pragma unroll
      for (int kk = 0; kk < KC; kk += 32) {
        v8bf lo = *reinterpret_cast<const v8bf*>(lrow + kk + 2 * g8);
        v8bf hi = *reinterpret_cast<const v8bf*>(lrow + kk + 2 * g8 + 8);
        v16bf a = __builtin_shufflevector(lo, hi, 0,1,2,3,4,5,6,7,8,9,10,11,12,13,14,15);
        int kabs = kc * KC + kk;
#pragma unroll
        for (int t = 0; t < 8; ++t)
          c[t] = WMMA_BF16(a, load_b_frag_bf(brow[t] + kabs), c[t]);
      }
      __syncthreads();
    }
#else
    for (int k0 = 0; k0 < HDIM; k0 += 32) {
      __builtin_prefetch(arow_g + k0 + 512, 0, 1);
      v16bf a = load_a_frag_bf(arow_g, g8, k0);
#pragma unroll
      for (int t = 0; t < 8; ++t)
        c[t] = WMMA_BF16(a, load_b_frag_bf(brow[t] + k0), c[t]);
    }
#endif
#pragma unroll
    for (int t = 0; t < 8; ++t) {
      int d = (grp * 8 + t) * 16 + ncol;
      float bias = b1[d];
      float w2d  = w2[d];
#pragma unroll
      for (int r = 0; r < 8; ++r) {
        float h = c[t][r] + bias;
        acc[r] += (h / (1.f + __expf(-h))) * w2d;   // silu * w2
      }
    }
  }
#pragma unroll
  for (int off = 8; off >= 1; off >>= 1) {
#pragma unroll
    for (int r = 0; r < 8; ++r) acc[r] += __shfl_xor(acc[r], off);
  }
  if ((lane & 15) == 0) {
    int mhi = ((lane >> 4) & 1) << 3;
    float bb = b2[0];
#pragma unroll
    for (int r = 0; r < 8; ++r) {
      float v = acc[r] + bb;
      difficulty[row0 + mhi + r] = 1.f / (1.f + __expf(-v));
    }
  }
}

// ----------------------- fallback f32-direct GEMM kernels ------------------
__global__ __launch_bounds__(128) void gate_logits_f32_kernel(
    const float* __restrict__ hidden, const float* __restrict__ gateW,
    float* __restrict__ logits) {
  int lane = threadIdx.x & 31;
  int wave = threadIdx.x >> 5;
  int row0 = (blockIdx.x * 4 + wave) * 16;
  const float* arow = hidden + (size_t)(row0 + (lane & 15)) * HDIM;
  int g16 = ((lane >> 4) & 1) << 4;
  int ncol = lane & 15;
  const float* b0p = gateW + (size_t)(ncol) * HDIM + g16;
  const float* b1p = gateW + (size_t)(16 + ncol) * HDIM + g16;
  const float* b2p = gateW + (size_t)(32 + ncol) * HDIM + g16;
  const float* b3p = gateW + (size_t)(48 + ncol) * HDIM + g16;
  v8f c0 = {}, c1 = {}, c2 = {}, c3 = {};
  for (int k0 = 0; k0 < HDIM; k0 += 32) {
    v16bf a = load_a_frag_f32(arow, lane, k0);
    c0 = WMMA_BF16(a, load_b_frag_f32(b0p + k0), c0);
    c1 = WMMA_BF16(a, load_b_frag_f32(b1p + k0), c1);
    c2 = WMMA_BF16(a, load_b_frag_f32(b2p + k0), c2);
    c3 = WMMA_BF16(a, load_b_frag_f32(b3p + k0), c3);
  }
  int mhi = ((lane >> 4) & 1) << 3;
#pragma unroll
  for (int r = 0; r < 8; ++r) {
    float* orow = logits + (size_t)(row0 + mhi + r) * NEXP + ncol;
    orow[0]  = c0[r];
    orow[16] = c1[r];
    orow[32] = c2[r];
    orow[48] = c3[r];
  }
}

__global__ __launch_bounds__(128) void difficulty_f32_kernel(
    const float* __restrict__ hidden, const float* __restrict__ w1,
    const float* __restrict__ b1, const float* __restrict__ w2,
    const float* __restrict__ b2, float* __restrict__ difficulty) {
  int lane = threadIdx.x & 31;
  int wave = threadIdx.x >> 5;
  int row0 = (blockIdx.x * 4 + wave) * 16;
  const float* arow = hidden + (size_t)(row0 + (lane & 15)) * HDIM;
  int g16 = ((lane >> 4) & 1) << 4;
  int ncol = lane & 15;
  float acc[8];
#pragma unroll
  for (int r = 0; r < 8; ++r) acc[r] = 0.f;
  for (int grp = 0; grp < 4; ++grp) {
    v8f c[8] = {};
    const float* brow[8];
#pragma unroll
    for (int t = 0; t < 8; ++t)
      brow[t] = w1 + (size_t)((grp * 8 + t) * 16 + ncol) * HDIM + g16;
    for (int k0 = 0; k0 < HDIM; k0 += 32) {
      v16bf a = load_a_frag_f32(arow, lane, k0);
#pragma unroll
      for (int t = 0; t < 8; ++t)
        c[t] = WMMA_BF16(a, load_b_frag_f32(brow[t] + k0), c[t]);
    }
#pragma unroll
    for (int t = 0; t < 8; ++t) {
      int d = (grp * 8 + t) * 16 + ncol;
      float bias = b1[d];
      float w2d  = w2[d];
#pragma unroll
      for (int r = 0; r < 8; ++r) {
        float h = c[t][r] + bias;
        acc[r] += (h / (1.f + __expf(-h))) * w2d;
      }
    }
  }
#pragma unroll
  for (int off = 8; off >= 1; off >>= 1) {
#pragma unroll
    for (int r = 0; r < 8; ++r) acc[r] += __shfl_xor(acc[r], off);
  }
  if ((lane & 15) == 0) {
    int mhi = ((lane >> 4) & 1) << 3;
    float bb = b2[0];
#pragma unroll
    for (int r = 0; r < 8; ++r) {
      float v = acc[r] + bb;
      difficulty[row0 + mhi + r] = 1.f / (1.f + __expf(-v));
    }
  }
}

// -------- per-token dynamic top-k + softmax (in-place logits->weights) -----
__global__ __launch_bounds__(256) void routing_kernel(
    float* __restrict__ inout, const float* __restrict__ difficulty,
    float* __restrict__ tpe, float* __restrict__ scal) {
  __shared__ float s_tpe[NEXP];
  __shared__ float s_k, s_ent;
  int tid = threadIdx.x;
  if (tid < NEXP) s_tpe[tid] = 0.f;
  if (tid == 0) { s_k = 0.f; s_ent = 0.f; }
  __syncthreads();

  int lane = tid & 31;
  int wave = tid >> 5;
  int tok = blockIdx.x * 8 + wave;
  float* row = inout + (size_t)tok * NEXP;
  float2 lg = *reinterpret_cast<const float2*>(row + 2 * lane);
  float l0 = lg.x, l1 = lg.y;

  float diff = difficulty[tok];
  float k_float = KMIN + diff * (KMAX - KMIN);
  int k = (int)fminf(fmaxf(rintf(k_float), KMIN), KMAX);

  const float NEG = -3.4e38f;
  bool sel0 = false, sel1 = false;
  float gmax = 0.f;
  for (int it = 0; it < k; ++it) {
    float v0 = sel0 ? NEG : l0;
    float v1 = sel1 ? NEG : l1;
    float v; int idx;
    if (v0 >= v1) { v = v0; idx = 2 * lane; } else { v = v1; idx = 2 * lane + 1; }
#pragma unroll
    for (int off = 16; off >= 1; off >>= 1) {
      float ov = __shfl_xor(v, off);
      int   oi = __shfl_xor(idx, off);
      if (ov > v || (ov == v && oi < idx)) { v = ov; idx = oi; }
    }
    if (it == 0) gmax = v;
    if (idx == 2 * lane) sel0 = true;
    else if (idx == 2 * lane + 1) sel1 = true;
  }

  float e0 = sel0 ? __expf(l0 - gmax) : 0.f;
  float e1 = sel1 ? __expf(l1 - gmax) : 0.f;
  float s = e0 + e1;
#pragma unroll
  for (int off = 16; off >= 1; off >>= 1) s += __shfl_xor(s, off);
  float inv = 1.f / s;
  float w0 = e0 * inv, w1v = e1 * inv;
  *reinterpret_cast<float2*>(row + 2 * lane) = make_float2(w0, w1v);

  if (w0 != 0.f)  atomicAdd(&s_tpe[2 * lane], w0);
  if (w1v != 0.f) atomicAdd(&s_tpe[2 * lane + 1], w1v);
  if (lane == 0) {
    float ent = diff * __logf(diff + 1e-8f) + (1.f - diff) * __logf(1.f - diff + 1e-8f);
    atomicAdd(&s_k, k_float);
    atomicAdd(&s_ent, ent);
  }
  __syncthreads();
  if (tid < NEXP)      atomicAdd(&tpe[tid], s_tpe[tid]);
  if (tid == NEXP)     atomicAdd(&scal[0], s_k);
  if (tid == NEXP + 1) atomicAdd(&scal[1], s_ent);
}

__global__ void init_kernel(float* acc) {
  int i = threadIdx.x;
  if (i < NEXP + 2) acc[i] = 0.f;
}

__global__ void finalize_kernel(const float* __restrict__ tpe,
                                const float* __restrict__ scal,
                                float* __restrict__ aux_out) {
  if (threadIdx.x == 0 && blockIdx.x == 0) {
    float mean = 0.f;
    for (int i = 0; i < NEXP; ++i) mean += tpe[i];
    mean /= (float)NEXP;
    float var = 0.f;
    for (int i = 0; i < NEXP; ++i) { float d = tpe[i] - mean; var += d * d; }
    var /= (float)(NEXP - 1);                         // ddof=1
    float balance = var / (mean + 1e-8f);
    float avg_k = scal[0] / (float)NTOK;
    float kp = fmaxf(BASEK - avg_k, 0.f); kp = kp * kp;
    float ent_mean = scal[1] / (float)NTOK;           // == -diff_entropy
    aux_out[0] = AUXW * (kp + balance) + ENTW * ent_mean;
  }
}

extern "C" void kernel_launch(void* const* d_in, const int* in_sizes, int n_in,
                              void* d_out, int out_size, void* d_ws, size_t ws_size,
                              hipStream_t stream) {
  (void)in_sizes; (void)n_in; (void)out_size;
  const float* hidden = (const float*)d_in[0];
  const float* gateW  = (const float*)d_in[1];
  const float* w1     = (const float*)d_in[2];
  const float* b1     = (const float*)d_in[3];
  const float* w2     = (const float*)d_in[4];
  const float* b2     = (const float*)d_in[5];

  float* out    = (float*)d_out;
  float* logits = out;                     // staged in-place, becomes weights
  float* aux    = out + (size_t)NTOK * NEXP;

  const size_t nH = (size_t)NTOK * HDIM, nW1 = (size_t)D1 * HDIM, nG = (size_t)NEXP * HDIM;
  const size_t need_bf = (nH + nW1 + nG) * sizeof(__bf16) + (NTOK + NEXP + 2) * sizeof(float);

  if (ws_size >= need_bf) {
    __bf16* hbf  = (__bf16*)d_ws;
    __bf16* wbf1 = hbf + nH;
    __bf16* gbf  = wbf1 + nW1;
    float* difficulty = (float*)(gbf + nG);
    float* tpe  = difficulty + NTOK;
    float* scal = tpe + NEXP;

    init_kernel<<<1, 128, 0, stream>>>(tpe);
    cvt_bf16_kernel<<<(int)(nH / 8 / 256), 256, 0, stream>>>(hidden, hbf, (int)(nH / 8));
    cvt_bf16_kernel<<<(int)(nW1 / 8 / 256), 256, 0, stream>>>(w1, wbf1, (int)(nW1 / 8));
    cvt_bf16_kernel<<<(int)(nG / 8 / 256), 256, 0, stream>>>(gateW, gbf, (int)(nG / 8));
    gate_logits_bf_kernel<<<NTOK / 64, 128, 0, stream>>>(hbf, gbf, logits);
    difficulty_bf_kernel<<<NTOK / 64, 128, 0, stream>>>(hbf, wbf1, b1, w2, b2, difficulty);
    routing_kernel<<<NTOK / 8, 256, 0, stream>>>(logits, difficulty, tpe, scal);
    finalize_kernel<<<1, 64, 0, stream>>>(tpe, scal, aux);
  } else {
    float* difficulty = (float*)d_ws;
    float* tpe  = difficulty + NTOK;
    float* scal = tpe + NEXP;
    init_kernel<<<1, 128, 0, stream>>>(tpe);
    gate_logits_f32_kernel<<<NTOK / 64, 128, 0, stream>>>(hidden, gateW, logits);
    difficulty_f32_kernel<<<NTOK / 64, 128, 0, stream>>>(hidden, w1, b1, w2, b2, difficulty);
    routing_kernel<<<NTOK / 8, 256, 0, stream>>>(logits, difficulty, tpe, scal);
    finalize_kernel<<<1, 64, 0, stream>>>(tpe, scal, aux);
  }
}
